// CDConv_28913719836781
// MI455X (gfx1250) — matrix operand order repaired and checked by reading, not compiled, not run on hardware
//
#include <hip/hip_runtime.h>

// ---------------------------------------------------------------------------
// Problem dims
// ---------------------------------------------------------------------------
#define BB    16
#define CC    128
#define HH    112
#define WW    112
#define HP    114            // padded (zero border) height for xT
#define WP    114            // padded width
#define OUTC  256
#define C2    256
#define PLANE (HH*WW)
#define NX    (BB*CC*HH*WW)  // 25,690,112

typedef __attribute__((ext_vector_type(16))) __bf16 v16bf;
typedef __attribute__((ext_vector_type(8)))  float  v8f;

// ---------------------------------------------------------------------------
// Zero-fill workspace region (for the padded xT border)
// ---------------------------------------------------------------------------
__global__ void zero16_kernel(uint4* __restrict__ p, long n16) {
    long i = (long)blockIdx.x * 256 + threadIdx.x;
    if (i < n16) {
        uint4 z; z.x = 0; z.y = 0; z.z = 0; z.w = 0;
        p[i] = z;
    }
}

// ---------------------------------------------------------------------------
// Transpose+convert x: [B][C][H][W] fp32 -> xT [B][HP][WP][C] bf16 (interior)
// One block = (b, h, 16-wide w tile). LDS tile padded (stride 17) so both
// phases are conflict-free; global reads and writes are coalesced.
// ---------------------------------------------------------------------------
__global__ __launch_bounds__(256)
void transpose_x_kernel(const float* __restrict__ x, __bf16* __restrict__ xT) {
    __shared__ float tile[CC * 17];
    int blk = blockIdx.x;                 // (b*H + h)*7 + wt
    int wt  = blk % 7;
    int bh  = blk / 7;
    int h   = bh % HH;
    int b   = bh / HH;
    int w0  = wt * 16;
    int tid = threadIdx.x;

    for (int i = tid; i < CC * 16; i += 256) {
        int c  = i >> 4;
        int wi = i & 15;
        tile[c * 17 + wi] = x[((size_t)(b * CC + c) * HH + h) * WW + w0 + wi];
    }
    __syncthreads();
    for (int i = tid; i < CC * 16; i += 256) {
        int wi = i >> 7;
        int c  = i & 127;
        xT[((size_t)(b * HP + h + 1) * WP + (w0 + wi + 1)) * CC + c]
            = (__bf16)tile[c * 17 + wi];
    }
}

// ---------------------------------------------------------------------------
// Repack weights into WMMA A-fragment order:  [kstep][mtile(16)][lane(32)][e(16)]
// A-fragment e->K mapping used here (contiguous, not the ISA table order; B uses the same):
//   lane l: m = l&15 ; kbase = (l<16)?0:16 ; K(e) = kbase + e
//  offset_w [256,128,3,3]: kglobal = tap*128 + c, tap = ky*3+kx
//  point_w  [256,128]    : kglobal = c
// ---------------------------------------------------------------------------
__global__ void cvt_w_kernel(const float* __restrict__ offw,
                             const float* __restrict__ pw,
                             __bf16* __restrict__ woF,
                             __bf16* __restrict__ pwF) {
    int idx = blockIdx.x * 256 + threadIdx.x;
    const int NWO = 36 * 16 * 32 * 16;     // 294,912
    const int NPW = 4 * 16 * 32 * 16;      // 32,768
    bool is_off = idx < NWO;
    if (idx >= NWO + NPW) return;
    int i = is_off ? idx : idx - NWO;

    int e  = i & 15;
    int l  = (i >> 4) & 31;
    int mt = (i >> 9) & 15;
    int ks = i >> 13;
    int m   = mt * 16 + (l & 15);
    int akb = (l < 16) ? 0 : 16;
    int K   = akb + e;
    int kg  = ks * 32 + K;

    if (is_off) {
        int tap = kg >> 7;       // 0..8
        int c   = kg & 127;
        woF[idx] = (__bf16)offw[m * 1152 + c * 9 + tap];
    } else {
        pwF[i] = (__bf16)pw[m * 128 + kg];
    }
}

// ---------------------------------------------------------------------------
// WMMA GEMM, fully LDS-free.  One workgroup per (b,h) row.
//   M = 256 out channels (8 waves x 2 M-tiles), N = 112 pixels (7 N-tiles),
//   K = KSTEPS*32.
// A fragments: contiguous 32B per lane from fragment-major weights (L2-hot).
// B fragments: contiguous 16 channels per lane from pixel-major activations;
//   offset conv taps become pointer shifts into the zero-padded xT (no
//   predication -> EXEC stays all-ones around every WMMA).
// B-fragment k order used here (contiguous, same as A packing): lane l: n = l&15, K(e) = ((l<16)?0:16)+e.
// C/D layout: lane l: n = l&15 ; vgpr r -> m = r + ((l<16)?0:8).
// ---------------------------------------------------------------------------
template<int KSTEPS, bool IS_OFFSET>
__global__ __launch_bounds__(256)
void gemm_bf16_kernel(const __bf16* __restrict__ Afrag,  // [KSTEPS][16][32][16]
                      const __bf16* __restrict__ BT,     // xT padded / smpT
                      const float*  __restrict__ bias,   // [256] (offset only)
                      float* __restrict__ Out)
{
    const int blk  = blockIdx.x;          // b*H + h
    const int b    = blk / HH;
    const int h    = blk % HH;
    const int lane = threadIdx.x & 31;
    const int wv   = threadIdx.x >> 5;    // 0..7
    const int nl   = lane & 15;
    const int klo  = (lane < 16) ? 0 : 16;

    v8f acc[2][7] = {};

    for (int ks = 0; ks < KSTEPS; ++ks) {
        // ---- A fragments: one v16bf (2x b128) each ----
        const __bf16* ap = Afrag + ((size_t)(ks * 16 + wv * 2) * 32 + lane) * 16;
        v16bf a0 = *(const v16bf*)ap;
        v16bf a1 = *(const v16bf*)(ap + 32 * 16);

        // ---- B row base for this k-step ----
        const __bf16* brow;
        if constexpr (IS_OFFSET) {
            const int tap   = ks >> 2;            // 0..8
            const int cbase = (ks & 3) << 5;      // 0,32,64,96
            const int dy    = tap / 3 - 1;
            const int dx    = tap % 3 - 1;
            brow = BT + ((size_t)(b * HP + (h + dy + 1)) * WP + (dx + 1)) * CC
                      + cbase + klo;
        } else {
            brow = BT + ((size_t)(b * HH + h) * WW) * CC + (ks << 5) + klo;
        }

        #pragma unroll
        for (int nt = 0; nt < 7; ++nt) {
            const int n = nt * 16 + nl;
            v16bf bf = *(const v16bf*)(brow + (size_t)n * CC);
            acc[0][nt] = __builtin_amdgcn_wmma_f32_16x16x32_bf16(
                false, a0, false, bf, (short)0, acc[0][nt], false, false);
            acc[1][nt] = __builtin_amdgcn_wmma_f32_16x16x32_bf16(
                false, a1, false, bf, (short)0, acc[1][nt], false, false);
        }
    }

    // ---- epilogue ----
    const int mofs = (lane < 16) ? 0 : 8;
    #pragma unroll
    for (int t = 0; t < 2; ++t) {
        const int mbase = (wv * 2 + t) * 16 + mofs;
        if constexpr (IS_OFFSET) {
            // tanh(acc + bias) -> pixel-major offT[p][256]; 8 contiguous floats/lane
            float bv[8];
            #pragma unroll
            for (int r = 0; r < 8; ++r) bv[r] = bias[mbase + r];
            #pragma unroll
            for (int nt = 0; nt < 7; ++nt) {
                const int n = nt * 16 + nl;
                float* dst = Out + ((size_t)(b * HH + h) * WW + n) * 256 + mbase;
                float4 o0, o1;
                o0.x = tanhf(acc[t][nt][0] + bv[0]);
                o0.y = tanhf(acc[t][nt][1] + bv[1]);
                o0.z = tanhf(acc[t][nt][2] + bv[2]);
                o0.w = tanhf(acc[t][nt][3] + bv[3]);
                o1.x = tanhf(acc[t][nt][4] + bv[4]);
                o1.y = tanhf(acc[t][nt][5] + bv[5]);
                o1.z = tanhf(acc[t][nt][6] + bv[6]);
                o1.w = tanhf(acc[t][nt][7] + bv[7]);
                *(float4*)dst       = o0;
                *(float4*)(dst + 4) = o1;
            }
        } else {
            // channel-major d_out [B][256][H][W]
            #pragma unroll
            for (int nt = 0; nt < 7; ++nt) {
                const int n = nt * 16 + nl;
                #pragma unroll
                for (int r = 0; r < 8; ++r) {
                    Out[((size_t)(b * 256 + mbase + r) * HH + h) * WW + n]
                        = acc[t][nt][r];
                }
            }
        }
    }
}

// ---------------------------------------------------------------------------
// Depthwise 3x3 conv, fp32 (memory-bound stencil, w-fastest for locality)
// ---------------------------------------------------------------------------
__global__ void dwconv_kernel(const float* __restrict__ x,
                              const float* __restrict__ wgt,
                              float* __restrict__ out) {
    int idx = blockIdx.x * 256 + threadIdx.x;
    if (idx >= NX) return;
    int w = idx % WW;
    int t = idx / WW;
    int h = t % HH;  t /= HH;
    int c = t % CC;
    int b = t / CC;
    const float* wp = wgt + c * 9;
    const float* xp = x + (size_t)(b * CC + c) * PLANE;
    float s = 0.f;
    #pragma unroll
    for (int ky = 0; ky < 3; ++ky) {
        int hy = h + ky - 1;
        if ((unsigned)hy >= (unsigned)HH) continue;
        #pragma unroll
        for (int kx = 0; kx < 3; ++kx) {
            int wx = w + kx - 1;
            if ((unsigned)wx >= (unsigned)WW) continue;
            s += xp[hy * WW + wx] * wp[ky * 3 + kx];
        }
    }
    out[idx] = s;
}

// ---------------------------------------------------------------------------
// Deformable bilinear sample (grid_sample, zeros pad, align_corners=False).
// Channel-fastest threading: offT reads and smpT writes fully coalesced;
// writes bf16 pixel-major for the downstream WMMA GEMM.
// ---------------------------------------------------------------------------
__global__ void sample_kernel(const float* __restrict__ offT,  // [p][256] fp32
                              const float* __restrict__ dw,    // [B][128][H][W] fp32
                              __bf16* __restrict__ smpT) {     // [p][128] bf16
    int idx = blockIdx.x * 256 + threadIdx.x;
    if (idx >= NX) return;
    int c = idx & 127;
    int p = idx >> 7;             // (b*H + h)*W + w
    int w = p % WW;
    int t = p / WW;
    int h = t % HH;
    int b = t / HH;

    const float gx = -1.f + (2.f / 111.f) * (float)w;
    const float gy = -1.f + (2.f / 111.f) * (float)h;
    const float ox = offT[(size_t)p * 256 + c];
    const float oy = offT[(size_t)p * 256 + 128 + c];

    float sx = fminf(fmaxf(gx + ox, -1.f), 1.f);
    float sy = fminf(fmaxf(gy + oy, -1.f), 1.f);
    float ix = (sx + 1.f) * (WW * 0.5f) - 0.5f;
    float iy = (sy + 1.f) * (HH * 0.5f) - 0.5f;
    float x0f = floorf(ix), y0f = floorf(iy);
    float wx = ix - x0f,   wy = iy - y0f;
    int x0 = (int)x0f, y0 = (int)y0f;

    const float* pp = dw + (size_t)(b * CC + c) * PLANE;
    float v00 = 0.f, v01 = 0.f, v10 = 0.f, v11 = 0.f;
    bool y0ok = (unsigned)y0 < (unsigned)HH, y1ok = (unsigned)(y0 + 1) < (unsigned)HH;
    bool x0ok = (unsigned)x0 < (unsigned)WW, x1ok = (unsigned)(x0 + 1) < (unsigned)WW;
    if (y0ok && x0ok) v00 = pp[y0 * WW + x0];
    if (y0ok && x1ok) v01 = pp[y0 * WW + x0 + 1];
    if (y1ok && x0ok) v10 = pp[(y0 + 1) * WW + x0];
    if (y1ok && x1ok) v11 = pp[(y0 + 1) * WW + x0 + 1];

    float v = v00 * (1.f - wx) * (1.f - wy) + v01 * wx * (1.f - wy)
            + v10 * (1.f - wx) * wy         + v11 * wx * wy;
    smpT[idx] = (__bf16)v;
}

// ---------------------------------------------------------------------------
// Host launcher
// ---------------------------------------------------------------------------
extern "C" void kernel_launch(void* const* d_in, const int* in_sizes, int n_in,
                              void* d_out, int out_size, void* d_ws, size_t ws_size,
                              hipStream_t stream) {
    const float* x        = (const float*)d_in[0];
    const float* depth_w  = (const float*)d_in[1];
    const float* point_w  = (const float*)d_in[2];
    const float* offset_w = (const float*)d_in[3];
    const float* offset_b = (const float*)d_in[4];
    float* out = (float*)d_out;

    // workspace carve-up (256B aligned)
    char*  ws = (char*)d_ws;
    size_t o  = 0;
    auto carve = [&](size_t bytes) -> void* {
        void* p = ws + o;
        o += (bytes + 255) & ~(size_t)255;
        return p;
    };
    const size_t XT_BYTES = (size_t)BB * HP * WP * CC * 2;      // 53,231,616
    __bf16* xT     = (__bf16*)carve(XT_BYTES);                  // padded pixel-major x
    __bf16* woF    = (__bf16*)carve((size_t)36 * 16 * 32 * 16 * 2);
    __bf16* pwF    = (__bf16*)carve((size_t)4 * 16 * 32 * 16 * 2);
    float*  offT   = (float*)carve((size_t)BB * PLANE * 256 * 4); // pixel-major offsets
    float*  dwbuf  = (float*)carve((size_t)NX * 4);
    __bf16* smpT   = (__bf16*)carve((size_t)NX * 2);              // pixel-major sampled

    const int TPB = 256;
    const long n16       = (long)(XT_BYTES / 16);
    const int nblk_zero  = (int)((n16 + TPB - 1) / TPB);
    const int nblk_tr    = BB * HH * 7;                   // 12,544
    const int nblk_w     = (36*16*32*16 + 4*16*32*16 + TPB - 1) / TPB;
    const int nblk_gemm  = BB * HH;                       // 1,792
    const int nblk_elem  = (NX + TPB - 1) / TPB;          // 100,352

    // 1) zero padded xT, then transpose interior; repack weights to frag order
    zero16_kernel<<<nblk_zero, TPB, 0, stream>>>((uint4*)xT, n16);
    transpose_x_kernel<<<nblk_tr, TPB, 0, stream>>>(x, xT);
    cvt_w_kernel<<<nblk_w, TPB, 0, stream>>>(offset_w, point_w, woF, pwF);

    // 2) offset conv 3x3 (implicit GEMM over 9 shifted taps) + bias + tanh
    gemm_bf16_kernel<36, true><<<nblk_gemm, TPB, 0, stream>>>(woF, xT, offset_b, offT);

    // 3) depthwise 3x3 (fp32)
    dwconv_kernel<<<nblk_elem, TPB, 0, stream>>>(x, depth_w, dwbuf);

    // 4) deformable bilinear sampling -> bf16 pixel-major
    sample_kernel<<<nblk_elem, TPB, 0, stream>>>(offT, dwbuf, smpT);

    // 5) pointwise 1x1 conv -> fp32 output (channel-major)
    gemm_bf16_kernel<4, false><<<nblk_gemm, TPB, 0, stream>>>(pwF, smpT, nullptr, out);

    (void)in_sizes; (void)n_in; (void)out_size; (void)ws_size;
}
